// RobustNeuralKalmanFilter_19026705121689
// MI455X (gfx1250) — matrix operand order repaired and checked by reading, not compiled, not run on hardware
//
#include <hip/hip_runtime.h>
#include <stdint.h>

// Problem dims
#define BATCH 8
#define SEQ   4096
#define HID   512
#define M_TOT (BATCH * SEQ)   // 32768
#define N_TOT (2 * HID)       // 1024
#define K_TOT HID             // 512

// GEMM tiling
#define BM 128
#define BN 128
#define BK 32
#define NK (K_TOT / BK)       // 16 k-steps (even)
#define LDT 40                // padded LDS row (bf16 elems) = 80B: 16B-aligned chunks, no bank conflicts

typedef __attribute__((ext_vector_type(16))) __bf16 v16bf;
typedef __attribute__((ext_vector_type(8)))  float  v8f;

__device__ __forceinline__ uint16_t f2bf(float f) {
    uint32_t u = __float_as_uint(f);
    u += 0x7FFFu + ((u >> 16) & 1u);     // round-to-nearest-even
    return (uint16_t)(u >> 16);
}
__device__ __forceinline__ float bf2f(uint16_t h) {
    return __uint_as_float(((uint32_t)h) << 16);
}

union FragU { uint4 u[2]; v16bf v; };

// CDNA5 async global->LDS copy (ASYNCcnt-tracked, no VGPR data path).
__device__ __forceinline__ void async_copy_b128(uint32_t lds_addr, const void* gaddr) {
    asm volatile("global_load_async_to_lds_b128 %0, %1, off"
                 :: "v"(lds_addr), "v"(gaddr) : "memory");
}
__device__ __forceinline__ uint32_t lds_addr_of(const void* p) {
    // low 32 bits of a flat shared-aperture address == LDS byte offset
    return (uint32_t)(uintptr_t)p;
}

// -------------------------------------------------------------------------
// fp32 -> bf16 bulk conversion (one pass; amortized over 8 N-block reuses)
// -------------------------------------------------------------------------
__global__ __launch_bounds__(256)
void convert_bf16_kernel(const float* __restrict__ in, uint16_t* __restrict__ out, int n8)
{
    int i = blockIdx.x * blockDim.x + threadIdx.x;
    if (i >= n8) return;
    const float4* p = (const float4*)in + (size_t)i * 2;
    float4 a = p[0], b = p[1];
    ushort4 ua{f2bf(a.x), f2bf(a.y), f2bf(a.z), f2bf(a.w)};
    ushort4 ub{f2bf(b.x), f2bf(b.y), f2bf(b.z), f2bf(b.w)};
    ushort4* q = (ushort4*)out + (size_t)i * 2;
    q[0] = ua;
    q[1] = ub;
}

// -------------------------------------------------------------------------
// Fused GEMM: noise[m][n] = softplus( sum_k X[m][k]*W[n][k] + bias[n] )
// bf16 inputs staged via double-buffered global_load_async_to_lds_b128,
// v_wmma_f32_16x16x32_bf16 compute, softplus epilogue, bf16 noise out.
// Rolled pipeline (pairs of K-steps) to keep VGPRs < 256, no spills.
// -------------------------------------------------------------------------
__global__ __launch_bounds__(256)
void gemm_softplus_kernel(const uint16_t* __restrict__ Xbf,
                          const uint16_t* __restrict__ Wbf,
                          const float* __restrict__ bias,
                          uint16_t* __restrict__ noise)
{
    __shared__ uint16_t lA[2 * BM * LDT];
    __shared__ uint16_t lB[2 * BN * LDT];

    const int tid  = threadIdx.x;
    const int wave = tid >> 5;
    const int lane = tid & 31;
    const int wm   = wave & 3;        // wave M index (4)
    const int wn   = wave >> 2;       // wave N index (2)
    const int half = lane >> 4;       // 0: lanes 0-15, 1: lanes 16-31
    const int l16  = lane & 15;

    const int n0 = blockIdx.x * BN;   // N fastest: X tile reused across 8 N-blocks via L2
    const int m0 = blockIdx.y * BM;

    // ---- per-thread staging addresses: 512 16B-chunks per tile, 2 per thread ----
    uint32_t lAoff[2][2], lBoff[2][2];
    const uint16_t* gA[2];            // advance by BK each issue
    const uint16_t* gB[2];
    #pragma unroll
    for (int i = 0; i < 2; i++) {
        int chunk = tid + i * 256;
        int row   = chunk >> 2;       // 4 chunks of 8 bf16 per 32-elem row
        int c8    = (chunk & 3) * 8;
        gA[i] = Xbf + (size_t)(m0 + row) * K_TOT + c8;
        gB[i] = Wbf + (size_t)(n0 + row) * K_TOT + c8;
        #pragma unroll
        for (int buf = 0; buf < 2; buf++) {
            lAoff[buf][i] = lds_addr_of(lA + buf * BM * LDT + row * LDT + c8);
            lBoff[buf][i] = lds_addr_of(lB + buf * BN * LDT + row * LDT + c8);
        }
    }

    v8f acc[2][4];
    #pragma unroll
    for (int i = 0; i < 2; i++)
        #pragma unroll
        for (int j = 0; j < 4; j++)
            acc[i][j] = v8f{0.f,0.f,0.f,0.f,0.f,0.f,0.f,0.f};

    // issue one K-step's staging (4 x b128 per thread) into buffer `buf`
    auto issue = [&](int buf) {
        #pragma unroll
        for (int i = 0; i < 2; i++) {
            async_copy_b128(lAoff[buf][i], gA[i]);
            async_copy_b128(lBoff[buf][i], gB[i]);
            gA[i] += BK;
            gB[i] += BK;
        }
    };

    // consume one staged K-step from buffer `buf`: 12 ds_load_b128 + 8 WMMA
    auto compute = [&](int buf) {
        const uint16_t* lAb = lA + buf * BM * LDT;
        const uint16_t* lBb = lB + buf * BN * LDT;
        v16bf afrag[2], bfrag[4];
        #pragma unroll
        for (int tm = 0; tm < 2; tm++) {
            const uint16_t* p = lAb + (wm * 32 + tm * 16 + l16) * LDT + half * 8;
            FragU f;
            f.u[0] = *(const uint4*)(p);
            f.u[1] = *(const uint4*)(p + 16);
            afrag[tm] = f.v;
        }
        #pragma unroll
        for (int tn = 0; tn < 4; tn++) {
            const uint16_t* p = lBb + (wn * 64 + tn * 16 + l16) * LDT + half * 8;
            FragU f;
            f.u[0] = *(const uint4*)(p);
            f.u[1] = *(const uint4*)(p + 16);
            bfrag[tn] = f.v;
        }
        #pragma unroll
        for (int tm = 0; tm < 2; tm++)
            #pragma unroll
            for (int tn = 0; tn < 4; tn++)
                acc[tm][tn] = __builtin_amdgcn_wmma_f32_16x16x32_bf16(
                    false, afrag[tm], false, bfrag[tn],
                    (short)0, acc[tm][tn], false, false);
    };

    // ---- software pipeline: prologue, rolled double-step body, 2-step tail ----
    issue(0);                                          // k-step 0 -> buf0

    #pragma unroll 1
    for (int kk2 = 0; kk2 < NK / 2 - 1; kk2++) {
        // even step: compute buf0, prefetch next into buf1
        issue(1);
        asm volatile("s_wait_asynccnt 0x4" ::: "memory");   // buf0 landed (in-order)
        __syncthreads();
        compute(0);
        __syncthreads();                                    // reads done before restage
        // odd step: compute buf1, prefetch next into buf0
        issue(0);
        asm volatile("s_wait_asynccnt 0x4" ::: "memory");
        __syncthreads();
        compute(1);
        __syncthreads();
    }
    // tail: k-steps NK-2 (buf0) and NK-1 (buf1)
    issue(1);
    asm volatile("s_wait_asynccnt 0x4" ::: "memory");
    __syncthreads();
    compute(0);
    __syncthreads();
    asm volatile("s_wait_asynccnt 0x0" ::: "memory");
    __syncthreads();
    compute(1);

    // ---- epilogue: bias + softplus, store bf16 ----
    // C layout: lane<16 -> (M=v, N=l16); lane>=16 -> (M=v+8, N=l16)
    #pragma unroll
    for (int tm = 0; tm < 2; tm++) {
        #pragma unroll
        for (int tn = 0; tn < 4; tn++) {
            int ng = n0 + wn * 64 + tn * 16 + l16;
            float bb = bias[ng];
            #pragma unroll
            for (int v = 0; v < 8; v++) {
                int mg = m0 + wm * 32 + tm * 16 + half * 8 + v;
                float x = acc[tm][tn][v] + bb;
                float sp = fmaxf(x, 0.0f) + __logf(1.0f + __expf(-fabsf(x)));
                noise[(size_t)mg * N_TOT + ng] = f2bf(sp);
            }
        }
    }
}

// -------------------------------------------------------------------------
// Kalman scan: one thread per (b,h) channel, sequential over S.
// Unrolled by 4 so z/q/r loads issue ahead of the dependent chain.
// -------------------------------------------------------------------------
__global__ __launch_bounds__(64)
void kalman_scan_kernel(const float* __restrict__ lstm,
                        const uint16_t* __restrict__ noise,
                        float* __restrict__ out)
{
    int c = blockIdx.x * blockDim.x + threadIdx.x;   // 0..4095
    if (c >= BATCH * HID) return;
    int b = c / HID;
    int h = c - b * HID;

    const float*    zp = lstm  + (size_t)b * SEQ * HID + h;
    const uint16_t* np = noise + (size_t)b * SEQ * N_TOT + h;
    float*          op = out   + (size_t)b * SEQ * HID + h;

    float state = zp[0];
    float P = 0.1f;

    for (int s = 0; s < SEQ; s += 4) {
        float z[4], q[4], r[4];
        #pragma unroll
        for (int j = 0; j < 4; j++) {
            z[j] = zp[(size_t)(s + j) * HID];
            q[j] = bf2f(np[(size_t)(s + j) * N_TOT]);
            r[j] = bf2f(np[(size_t)(s + j) * N_TOT + HID]);
        }
        #pragma unroll
        for (int j = 0; j < 4; j++) {
            float Pp = P + q[j];
            float K  = Pp / (Pp + r[j] + 1e-6f);
            state    = state + K * (z[j] - state);
            P        = (1.0f - K) * Pp;
            op[(size_t)(s + j) * HID] = state;
        }
    }
}

// -------------------------------------------------------------------------
extern "C" void kernel_launch(void* const* d_in, const int* in_sizes, int n_in,
                              void* d_out, int out_size, void* d_ws, size_t ws_size,
                              hipStream_t stream)
{
    const float* lstm = (const float*)d_in[0];   // (8, 4096, 512) fp32
    const float* W    = (const float*)d_in[1];   // (1024, 512)    fp32
    const float* bias = (const float*)d_in[2];   // (1024,)        fp32
    float* out = (float*)d_out;                  // (8, 4096, 512) fp32

    // workspace layout: Xbf16 32MiB | Wbf16 1MiB | noise bf16 64MiB
    uint16_t* xbf   = (uint16_t*)d_ws;
    uint16_t* wbf   = xbf + (size_t)M_TOT * K_TOT;
    uint16_t* noise = wbf + (size_t)N_TOT * K_TOT;

    int nx8 = (M_TOT * K_TOT) / 8;
    convert_bf16_kernel<<<(nx8 + 255) / 256, 256, 0, stream>>>(lstm, xbf, nx8);
    int nw8 = (N_TOT * K_TOT) / 8;
    convert_bf16_kernel<<<(nw8 + 255) / 256, 256, 0, stream>>>(W, wbf, nw8);

    dim3 grid(N_TOT / BN, M_TOT / BM);           // (8, 256), N fastest for X reuse
    gemm_softplus_kernel<<<grid, 256, 0, stream>>>(xbf, wbf, bias, noise);

    kalman_scan_kernel<<<(BATCH * HID + 63) / 64, 64, 0, stream>>>(lstm, noise, out);
}
